// AttentionBlock_16767552324152
// MI455X (gfx1250) — matrix operand order repaired and checked by reading, not compiled, not run on hardware
//
#include <hip/hip_runtime.h>
#include <hip/hip_bf16.h>

#define B_   8
#define C_   64
#define IC_  32
#define N_   4096      // H*W
#define NKV_ 1024      // (H/2)*(W/2)
#define NPIX (B_*N_)   // 32768

typedef __attribute__((ext_vector_type(16))) _Float16 v16h;
typedef __attribute__((ext_vector_type(8)))  float    v8f;

union AF { v16h v; unsigned u[8]; _Float16 h[16]; };

// ---------------------------------------------------------------------------
// DPP16 butterfly reduction within a 16-lane half-wave (no LDS, pure VALU).
// quad_perm(1,0,3,2)=0xB1, quad_perm(2,3,0,1)=0x4E,
// row_half_mirror=0x141, row_mirror=0x140
// ---------------------------------------------------------------------------
template <int CTRL>
__device__ __forceinline__ float dpp_xfer(float v) {
  int t = __builtin_amdgcn_update_dpp(0, __float_as_int(v), CTRL, 0xf, 0xf, true);
  return __int_as_float(t);
}
__device__ __forceinline__ float rowmax16(float v) {
  v = fmaxf(v, dpp_xfer<0x0B1>(v));
  v = fmaxf(v, dpp_xfer<0x04E>(v));
  v = fmaxf(v, dpp_xfer<0x141>(v));
  v = fmaxf(v, dpp_xfer<0x140>(v));
  return v;
}
__device__ __forceinline__ float rowsum16(float v) {
  v += dpp_xfer<0x0B1>(v);
  v += dpp_xfer<0x04E>(v);
  v += dpp_xfer<0x141>(v);
  v += dpp_xfer<0x140>(v);
  return v;
}

// ---------------------------------------------------------------------------
// theta projection: theta_x[b,ic,n] -> f16 thetaT[b][n][ic]
// ---------------------------------------------------------------------------
__global__ __launch_bounds__(256)
void theta_kernel(const float* __restrict__ x,
                  const float* __restrict__ tw, const float* __restrict__ tb,
                  _Float16* __restrict__ thetaT) {
  __shared__ float w[IC_*C_];
  __shared__ float bb[IC_];
  for (int i = threadIdx.x; i < IC_*C_; i += 256) w[i] = tw[i];
  if (threadIdx.x < IC_) bb[threadIdx.x] = tb[threadIdx.x];
  __syncthreads();

  int pix = blockIdx.x * 256 + threadIdx.x;      // 0..32767
  int b = pix >> 12, n = pix & (N_ - 1);
  const float* xp = x + (size_t)b * C_ * N_ + n;

  float acc[IC_];
#pragma unroll
  for (int ic = 0; ic < IC_; ++ic) acc[ic] = bb[ic];
  for (int c = 0; c < C_; ++c) {
    float xv = xp[(size_t)c * N_];
#pragma unroll
    for (int ic = 0; ic < IC_; ++ic) acc[ic] = fmaf(w[ic*C_ + c], xv, acc[ic]);
  }
  _Float16* op = thetaT + (size_t)pix * IC_;
#pragma unroll
  for (int ic = 0; ic < IC_; ++ic) op[ic] = (_Float16)acc[ic];
}

// ---------------------------------------------------------------------------
// pooled projections: kvT[b][k][ic] = max4(phi)+max4(lf) ; g2[b][ic][k] = max4(g)
// ---------------------------------------------------------------------------
__global__ __launch_bounds__(256)
void pooled_kernel(const float* __restrict__ x, const float* __restrict__ lfi,
                   const float* __restrict__ gw, const float* __restrict__ gb,
                   const float* __restrict__ pw, const float* __restrict__ pb,
                   const float* __restrict__ hw, const float* __restrict__ hb,
                   _Float16* __restrict__ kvT, _Float16* __restrict__ g2) {
  __shared__ float sgw[IC_*C_], spw[IC_*C_], shw[IC_*IC_];
  __shared__ float sgb[IC_], spb[IC_], shb[IC_];
  for (int i = threadIdx.x; i < IC_*C_; i += 256) { sgw[i] = gw[i]; spw[i] = pw[i]; }
  for (int i = threadIdx.x; i < IC_*IC_; i += 256) shw[i] = hw[i];
  if (threadIdx.x < IC_) {
    sgb[threadIdx.x] = gb[threadIdx.x];
    spb[threadIdx.x] = pb[threadIdx.x];
    shb[threadIdx.x] = hb[threadIdx.x];
  }
  __syncthreads();

  int idx = blockIdx.x * 256 + threadIdx.x;      // 0..8191
  int b = idx >> 10, k = idx & (NKV_ - 1);
  int ph = k >> 5, pwi = k & 31;

  float pmax[IC_], gmax[IC_], lmax[IC_];
#pragma unroll
  for (int ic = 0; ic < IC_; ++ic) { pmax[ic] = -1e30f; gmax[ic] = -1e30f; lmax[ic] = -1e30f; }

#pragma unroll
  for (int p = 0; p < 4; ++p) {
    int n = (2*ph + (p >> 1)) * 64 + 2*pwi + (p & 1);
    {
      const float* xp = x + (size_t)b * C_ * N_ + n;
      float pa[IC_], ga[IC_];
#pragma unroll
      for (int ic = 0; ic < IC_; ++ic) { pa[ic] = spb[ic]; ga[ic] = sgb[ic]; }
      for (int c = 0; c < C_; ++c) {
        float xv = xp[(size_t)c * N_];
#pragma unroll
        for (int ic = 0; ic < IC_; ++ic) {
          pa[ic] = fmaf(spw[ic*C_ + c], xv, pa[ic]);
          ga[ic] = fmaf(sgw[ic*C_ + c], xv, ga[ic]);
        }
      }
#pragma unroll
      for (int ic = 0; ic < IC_; ++ic) {
        pmax[ic] = fmaxf(pmax[ic], pa[ic]);
        gmax[ic] = fmaxf(gmax[ic], ga[ic]);
      }
    }
    {
      const float* lp = lfi + (size_t)b * IC_ * N_ + n;
      float la[IC_];
#pragma unroll
      for (int ic = 0; ic < IC_; ++ic) la[ic] = shb[ic];
      for (int c = 0; c < IC_; ++c) {
        float lv = lp[(size_t)c * N_];
#pragma unroll
        for (int ic = 0; ic < IC_; ++ic) la[ic] = fmaf(shw[ic*IC_ + c], lv, la[ic]);
      }
#pragma unroll
      for (int ic = 0; ic < IC_; ++ic) lmax[ic] = fmaxf(lmax[ic], la[ic]);
    }
  }
  _Float16* kvp = kvT + (size_t)idx * IC_;
#pragma unroll
  for (int ic = 0; ic < IC_; ++ic) kvp[ic] = (_Float16)(pmax[ic] + lmax[ic]);
  _Float16* gp = g2 + (size_t)b * IC_ * NKV_ + k;
#pragma unroll
  for (int ic = 0; ic < IC_; ++ic) gp[(size_t)ic * NKV_] = (_Float16)gmax[ic];
}

// ---------------------------------------------------------------------------
// fused flash attention: scores -> online softmax -> y  (all WMMA f16->f32)
// one wave per 16-row M tile; 8 waves / block
// ---------------------------------------------------------------------------
__global__ __launch_bounds__(256)
void attn_kernel(const _Float16* __restrict__ thetaT,
                 const _Float16* __restrict__ kvT,
                 const _Float16* __restrict__ g2,
                 float* __restrict__ yT) {
  __shared__ _Float16 P[8][16][32];      // per-wave probs tile (A2 staging)

  int tid  = threadIdx.x;
  int wv   = tid >> 5;
  int lane = tid & 31;
  int half = lane >> 4;
  int l16  = lane & 15;
  int b  = blockIdx.x >> 5;
  int tb = blockIdx.x & 31;
  int m0 = tb * 128 + wv * 16;

  // A fragment: theta^T rows m0..m0+15, K = ic (16-bit A 16x32 layout)
  AF a;
  {
    const _Float16* base = thetaT + (size_t)(b * N_ + m0 + l16) * IC_;
    int kh = half * 16;
#pragma unroll
    for (int v = 0; v < 8; ++v) {
      int K = 2*v + kh;
      a.u[v] = *(const unsigned*)(base + K);
    }
  }

  float m_run[8], l_run[8];
#pragma unroll
  for (int r = 0; r < 8; ++r) { m_run[r] = -1e30f; l_run[r] = 0.0f; }
  v8f o0 = {}; v8f o1 = {};

  for (int kc = 0; kc < NKV_; kc += 32) {
    // B fragments for two 16-col score tiles (16-bit B 32x16 layout)
    AF bk0, bk1;
    {
      int Kh = half * 16;
      const _Float16* k0 = kvT + (size_t)(b * NKV_ + kc + l16) * IC_ + Kh;
      const _Float16* k1 = kvT + (size_t)(b * NKV_ + kc + 16 + l16) * IC_ + Kh;
#pragma unroll
      for (int v = 0; v < 8; ++v) {
        bk0.u[v] = *(const unsigned*)(k0 + 2*v);
        bk1.u[v] = *(const unsigned*)(k1 + 2*v);
      }
    }
    v8f zero = {};
    v8f s0 = __builtin_amdgcn_wmma_f32_16x16x32_f16(false, a.v, false, bk0.v,
                                                    (short)0, zero, false, false);
    v8f s1 = __builtin_amdgcn_wmma_f32_16x16x32_f16(false, a.v, false, bk1.v,
                                                    (short)0, zero, false, false);

    // online softmax per row: row M = r + 8*half lives across one 16-lane
    // half-wave == one DPP16 row, so reductions are pure-VALU DPP butterflies
#pragma unroll
    for (int r = 0; r < 8; ++r) {
      float rm = rowmax16(fmaxf(s0[r], s1[r]));
      float mnew = fmaxf(m_run[r], rm);
      float sc = __expf(m_run[r] - mnew);
      float p0 = __expf(s0[r] - mnew);
      float p1 = __expf(s1[r] - mnew);
      float rs = rowsum16(p0 + p1);
      l_run[r] = l_run[r] * sc + rs;
      m_run[r] = mnew;
      o0[r] *= sc;
      o1[r] *= sc;
      P[wv][r + half*8][l16]      = (_Float16)p0;
      P[wv][r + half*8][16 + l16] = (_Float16)p1;
    }

    // re-stripe probs into A-fragment layout (same-wave DS is in-order)
    asm volatile("s_wait_dscnt 0" ::: "memory");
    AF a2;
    {
      int kh = half * 16;
#pragma unroll
      for (int v = 0; v < 8; ++v) {
        int K = 2*v + kh;
        a2.u[v] = *(const unsigned*)(&P[wv][l16][K]);
      }
    }

    // B fragments of g^T: K = pooled index (contiguous in g2), N = ic
    AF bg0, bg1;
    {
      const _Float16* g0 = g2 + (size_t)b * IC_ * NKV_ + (size_t)l16 * NKV_ + kc + half*16;
      const _Float16* g1 = g2 + (size_t)b * IC_ * NKV_ + (size_t)(16 + l16) * NKV_ + kc + half*16;
#pragma unroll
      for (int v = 0; v < 8; ++v) {
        bg0.u[v] = *(const unsigned*)(g0 + 2*v);
        bg1.u[v] = *(const unsigned*)(g1 + 2*v);
      }
    }
    o0 = __builtin_amdgcn_wmma_f32_16x16x32_f16(false, a2.v, false, bg0.v,
                                                (short)0, o0, false, false);
    o1 = __builtin_amdgcn_wmma_f32_16x16x32_f16(false, a2.v, false, bg1.v,
                                                (short)0, o1, false, false);
    asm volatile("" ::: "memory");
  }

  // normalize by softmax denominator and store y^T [b][n][ic] (f32)
  float* outp = yT + (size_t)(b * N_ + m0) * IC_;
#pragma unroll
  for (int r = 0; r < 8; ++r) {
    int M = r + half * 8;
    float inv = 1.0f / l_run[r];
    outp[(size_t)M * IC_ + l16]      = o0[r] * inv;
    outp[(size_t)M * IC_ + 16 + l16] = o1[r] * inv;
  }
}

// ---------------------------------------------------------------------------
// BatchNorm statistics: one block per output channel, wy computed on the fly
// ---------------------------------------------------------------------------
__global__ __launch_bounds__(256)
void stats_kernel(const float* __restrict__ yT, const float* __restrict__ ww,
                  const float* __restrict__ wb, float* __restrict__ stats) {
  int c = blockIdx.x;
  __shared__ float red[512];
  float wrow[IC_];
#pragma unroll
  for (int ic = 0; ic < IC_; ++ic) wrow[ic] = ww[c*IC_ + ic];
  float bias = wb[c];
  float s = 0.0f, ss = 0.0f;
  for (int p = threadIdx.x; p < NPIX; p += 256) {
    const float* yp = yT + (size_t)p * IC_;
    float v = bias;
#pragma unroll
    for (int ic = 0; ic < IC_; ++ic) v = fmaf(wrow[ic], yp[ic], v);
    s += v; ss += v * v;
  }
  red[threadIdx.x] = s; red[256 + threadIdx.x] = ss;
  __syncthreads();
  for (int off = 128; off > 0; off >>= 1) {
    if (threadIdx.x < off) {
      red[threadIdx.x]       += red[threadIdx.x + off];
      red[256 + threadIdx.x] += red[256 + threadIdx.x + off];
    }
    __syncthreads();
  }
  if (threadIdx.x == 0) { stats[2*c] = red[0]; stats[2*c + 1] = red[256]; }
}

// ---------------------------------------------------------------------------
// apply: wy = W*y + b, BN(scale/shift), + residual x
// ---------------------------------------------------------------------------
__global__ __launch_bounds__(256)
void apply_kernel(const float* __restrict__ yT, const float* __restrict__ x,
                  const float* __restrict__ ww, const float* __restrict__ wb,
                  const float* __restrict__ gamma, const float* __restrict__ beta,
                  const float* __restrict__ stats, float* __restrict__ out) {
  __shared__ float sw[C_*IC_];
  __shared__ float ssc[C_], ssh[C_], sbias[C_];
  for (int i = threadIdx.x; i < C_*IC_; i += 256) sw[i] = ww[i];
  if (threadIdx.x < C_) {
    int c = threadIdx.x;
    float cnt  = (float)NPIX;
    float mean = stats[2*c] / cnt;
    float var  = stats[2*c + 1] / cnt - mean * mean;
    float scl  = gamma[c] * rsqrtf(var + 1e-5f);
    ssc[c]   = scl;
    ssh[c]   = beta[c] - mean * scl;
    sbias[c] = wb[c];
  }
  __syncthreads();

  int pix = blockIdx.x * 256 + threadIdx.x;
  int b = pix >> 12, n = pix & (N_ - 1);
  const float* yp = yT + (size_t)pix * IC_;
  float y[IC_];
#pragma unroll
  for (int ic = 0; ic < IC_; ++ic) y[ic] = yp[ic];
  float*       op = out + (size_t)b * C_ * N_ + n;
  const float* xp = x   + (size_t)b * C_ * N_ + n;
  for (int c = 0; c < C_; ++c) {
    float v = sbias[c];
#pragma unroll
    for (int ic = 0; ic < IC_; ++ic) v = fmaf(sw[c*IC_ + ic], y[ic], v);
    op[(size_t)c * N_] = v * ssc[c] + ssh[c] + xp[(size_t)c * N_];
  }
}

// ---------------------------------------------------------------------------
extern "C" void kernel_launch(void* const* d_in, const int* in_sizes, int n_in,
                              void* d_out, int out_size, void* d_ws, size_t ws_size,
                              hipStream_t stream) {
  const float* x       = (const float*)d_in[0];
  const float* lfi     = (const float*)d_in[1];
  const float* g_w     = (const float*)d_in[2];
  const float* g_b     = (const float*)d_in[3];
  const float* theta_w = (const float*)d_in[4];
  const float* theta_b = (const float*)d_in[5];
  const float* phi_w   = (const float*)d_in[6];
  const float* phi_b   = (const float*)d_in[7];
  const float* w_w     = (const float*)d_in[8];
  const float* w_b     = (const float*)d_in[9];
  const float* bn_g    = (const float*)d_in[10];
  const float* bn_b    = (const float*)d_in[11];
  const float* hcc_w   = (const float*)d_in[12];
  const float* hcc_b   = (const float*)d_in[13];

  _Float16* thetaT = (_Float16*)d_ws;                        // 2 MB
  _Float16* kvT    = thetaT + (size_t)B_ * N_ * IC_;         // 0.5 MB
  _Float16* g2     = kvT    + (size_t)B_ * NKV_ * IC_;       // 0.5 MB
  float*    yT     = (float*)(g2 + (size_t)B_ * NKV_ * IC_); // 4 MB
  float*    stats  = yT + (size_t)B_ * N_ * IC_;             // 512 B
  float*    out    = (float*)d_out;

  theta_kernel <<<NPIX/256, 256, 0, stream>>>(x, theta_w, theta_b, thetaT);
  pooled_kernel<<<(B_*NKV_)/256, 256, 0, stream>>>(x, lfi, g_w, g_b, phi_w, phi_b,
                                                   hcc_w, hcc_b, kvT, g2);
  attn_kernel  <<<B_*(N_/128), 256, 0, stream>>>(thetaT, kvT, g2, yT);
  stats_kernel <<<C_, 256, 0, stream>>>(yT, w_w, w_b, stats);
  apply_kernel <<<NPIX/256, 256, 0, stream>>>(yT, x, w_w, w_b, bn_g, bn_b, stats, out);
}
